// BaseLayer_10514079940683
// MI455X (gfx1250) — matrix operand order repaired and checked by reading, not compiled, not run on hardware
//
#include <hip/hip_runtime.h>
#include <hip/hip_bf16.h>
#include <math.h>

// ---------------------------------------------------------------------------
// Problem constants (match reference setup_inputs)
// ---------------------------------------------------------------------------
#define DD   1024          // model dim d
#define FF   4096          // ffn dim f
#define TT   2048          // tokens (B*S = 2*1024)
#define NE   8             // experts
#define MAXROWS 136        // max 16-token tile rows: T/16 + E
#define LISTPAD (MAXROWS*16)   // 2176 padded list slots
#define NBLK 4             // N-tiles (16 cols each) per wave => 64 cols/wave

typedef __attribute__((ext_vector_type(16))) __bf16 v16bf;
typedef __attribute__((ext_vector_type(8)))  __bf16 v8bf;
typedef __attribute__((ext_vector_type(8)))  float  v8f;

// ---------------------------------------------------------------------------
// Kernel 1: per-token routing + LayerNorm + expert-affine, store h as bf16
//   one 256-thread block per token
// ---------------------------------------------------------------------------
__global__ __launch_bounds__(256) void prep_kernel(
    const float* __restrict__ x, const float* __restrict__ cent,
    const float* __restrict__ gamma, const float* __restrict__ beta,
    __bf16* __restrict__ hbuf, int* __restrict__ eid_out,
    float* __restrict__ alpha_out)
{
    const int t   = blockIdx.x;
    const int tid = threadIdx.x;
    const float* xr = x + (long)t * DD;

    float xv[4];
    float vals[10];                       // [0]=sum [1]=sumsq [2..9]=aff
    #pragma unroll
    for (int q = 0; q < 10; ++q) vals[q] = 0.f;

    #pragma unroll
    for (int i = 0; i < 4; ++i) {
        const int j = tid + i * 256;
        const float v = xr[j];
        xv[i] = v;
        vals[0] += v;
        vals[1] += v * v;
        #pragma unroll
        for (int e = 0; e < NE; ++e) vals[2 + e] += v * cent[e * DD + j];
    }

    __shared__ float red[256];
    __shared__ float res[10];
    for (int q = 0; q < 10; ++q) {
        red[tid] = vals[q];
        __syncthreads();
        for (int o = 128; o > 0; o >>= 1) {
            if (tid < o) red[tid] += red[tid + o];
            __syncthreads();
        }
        if (tid == 0) res[q] = red[0];
        __syncthreads();
    }

    __shared__ float bmu, brstd;
    __shared__ int   be;
    if (tid == 0) {
        const float mu   = res[0] * (1.0f / DD);
        const float var  = res[1] * (1.0f / DD) - mu * mu;
        const float rstd = rsqrtf(var + 1e-5f);
        int best = 0; float bv = res[2];
        for (int e = 1; e < NE; ++e)
            if (res[2 + e] > bv) { bv = res[2 + e]; best = e; }
        eid_out[t]   = best;
        alpha_out[t] = 1.0f / (1.0f + expf(-bv));
        bmu = mu; brstd = rstd; be = best;
    }
    __syncthreads();

    const float* g  = gamma + (long)be * DD;
    const float* bt = beta  + (long)be * DD;
    #pragma unroll
    for (int i = 0; i < 4; ++i) {
        const int j = tid + i * 256;
        const float xh = (xv[i] - bmu) * brstd;
        hbuf[(long)t * DD + j] = (__bf16)(xh * g[j] + bt[j]);
    }
}

// ---------------------------------------------------------------------------
// Kernel 2: deterministic per-expert token lists + tile descriptors (1 block)
// ---------------------------------------------------------------------------
__global__ __launch_bounds__(256) void scan_kernel(
    const int* __restrict__ eid, int* __restrict__ list,
    int* __restrict__ desc_e, int* __restrict__ desc_s,
    __bf16* __restrict__ zerobuf)
{
    const int tid = threadIdx.x;
    __shared__ int cnt[NE];
    if (tid < NE) cnt[tid] = 0;
    __syncthreads();
    for (int t = tid; t < TT; t += 256) atomicAdd(&cnt[eid[t]], 1);
    for (int i = tid; i < FF;      i += 256) zerobuf[i] = (__bf16)0.0f;
    for (int i = tid; i < LISTPAD; i += 256) list[i]    = -1;
    __syncthreads();

    if (tid == 0) {
        int off[NE + 1]; off[0] = 0;
        for (int e = 0; e < NE; ++e)
            off[e + 1] = off[e] + ((cnt[e] + 15) / 16) * 16;
        int cur[NE];
        for (int e = 0; e < NE; ++e) cur[e] = off[e];
        for (int t = 0; t < TT; ++t) {            // stable, deterministic
            const int e = eid[t];
            list[cur[e]++] = t;
        }
        int rows = 0;
        for (int e = 0; e < NE; ++e) {
            const int nr = (cnt[e] + 15) / 16;
            for (int r = 0; r < nr; ++r) {
                desc_e[rows] = e;
                desc_s[rows] = off[e] + r * 16;
                ++rows;
            }
        }
        for (; rows < MAXROWS; ++rows) desc_e[rows] = -1;
    }
}

// ---------------------------------------------------------------------------
// Kernel 3: transpose-convert f32 [z][R][C] -> bf16 [z][C][R]
// ---------------------------------------------------------------------------
__global__ __launch_bounds__(256) void tconv_kernel(
    const float* __restrict__ src, __bf16* __restrict__ dst, int R, int C)
{
    __shared__ float tile[32][33];
    const long base = (long)blockIdx.z * R * C;
    const int  c0 = blockIdx.x * 32, r0 = blockIdx.y * 32;
    const int  tx = threadIdx.x & 31, ty = threadIdx.x >> 5;   // 32x8

    const float* s = src + base;
    __bf16*      d = dst + base;
    #pragma unroll
    for (int rr = 0; rr < 32; rr += 8)
        tile[ty + rr][tx] = s[(long)(r0 + ty + rr) * C + (c0 + tx)];
    __syncthreads();
    #pragma unroll
    for (int rr = 0; rr < 32; rr += 8)
        d[(long)(c0 + ty + rr) * R + (r0 + tx)] = (__bf16)tile[tx][ty + rr];
}

// ---------------------------------------------------------------------------
// WMMA fragment helpers (bf16 16x16x32, wave32 layouts per CDNA5 ISA 7.12.2)
//   A lane (L): row M=L&15, K in {kb..kb+7, kb+16..kb+23}, kb = (L>=16)*8
//   B lane:     col N=L&15, K in {kb..kb+7, kb+16..kb+23}, kb = (L>=16)*8
// ---------------------------------------------------------------------------
__device__ __forceinline__ v16bf load_frag_A(const __bf16* p, int k0, int kbA)
{
    const v8bf lo = *(const v8bf*)(p + k0 + kbA);
    const v8bf hi = *(const v8bf*)(p + k0 + kbA + 16);
    return __builtin_shufflevector(lo, hi, 0,1,2,3,4,5,6,7,8,9,10,11,12,13,14,15);
}
__device__ __forceinline__ v16bf load_frag_B(const __bf16* p, int k0, int kbB)
{
    const v8bf lo = *(const v8bf*)(p + k0 + kbB / 2);
    const v8bf hi = *(const v8bf*)(p + k0 + kbB / 2 + 16);
    return __builtin_shufflevector(lo, hi, 0,1,2,3,4,5,6,7,8,9,10,11,12,13,14,15);
}

// ---------------------------------------------------------------------------
// Kernel 4: grouped GEMM1  H = relu(h @ w1[e] + b1[e]),  K = d
//   one wave computes a 16x64 tile (4 accumulators, A-fragment reuse)
//   grid: (FF/(16*NBLK*8) = 8, MAXROWS); block 256 = 8 waves
// ---------------------------------------------------------------------------
__global__ __launch_bounds__(256) void gemm1_kernel(
    const __bf16* __restrict__ hbuf, const __bf16* __restrict__ w1t,
    const float* __restrict__ b1, const int* __restrict__ list,
    const int* __restrict__ desc_e, const int* __restrict__ desc_s,
    const __bf16* __restrict__ zerobuf, __bf16* __restrict__ Hbuf)
{
    const int row = blockIdx.y;
    const int e   = desc_e[row];
    if (e < 0) return;
    const int rs   = desc_s[row];
    const int wv   = threadIdx.x >> 5;
    const int lane = threadIdx.x & 31;
    const int ncol0 = (blockIdx.x * 8 + wv) * (16 * NBLK);
    const int m    = lane & 15;
    const int half = lane >> 4;
    const int kbA  = half * 8;
    const int kbB  = half * 16;

    const int t = list[rs + m];
    const __bf16* aptr = (t >= 0) ? (hbuf + (long)t * DD) : zerobuf;
    const __bf16* wbase = w1t + (long)e * DD * FF;
    const __bf16* bcol[NBLK];
    #pragma unroll
    for (int j = 0; j < NBLK; ++j)
        bcol[j] = wbase + (long)(ncol0 + j * 16 + m) * DD;

    v8f cacc[NBLK];
    #pragma unroll
    for (int j = 0; j < NBLK; ++j) cacc[j] = {};

    #pragma unroll 2
    for (int k0 = 0; k0 < DD; k0 += 32) {
        const v16bf a = load_frag_A(aptr, k0, kbA);
        #pragma unroll
        for (int j = 0; j < NBLK; ++j) {
            const v16bf b = load_frag_B(bcol[j], k0, kbB);
            cacc[j] = __builtin_amdgcn_wmma_f32_16x16x32_bf16(
                false, a, false, b, (short)0, cacc[j], false, false);
        }
    }

    #pragma unroll
    for (int j = 0; j < NBLK; ++j) {
        const int   col  = ncol0 + j * 16 + m;
        const float bias = b1[(long)e * FF + col];
        #pragma unroll
        for (int r = 0; r < 8; ++r) {
            const int M = r + half * 8;
            float h = cacc[j][r] + bias;
            h = h > 0.0f ? h : 0.0f;
            Hbuf[(long)(rs + M) * FF + col] = (__bf16)h;
        }
    }
}

// ---------------------------------------------------------------------------
// Kernel 5: grouped GEMM2 + gated residual scatter,  K = f
//   out[t] = x[t] + alpha[t] * (H @ w2[e] + b2[e])
//   one wave computes a 16x64 tile; grid: (DD/(16*NBLK*8) = 2, MAXROWS)
// ---------------------------------------------------------------------------
__global__ __launch_bounds__(256) void gemm2_kernel(
    const __bf16* __restrict__ Hbuf, const __bf16* __restrict__ w2t,
    const float* __restrict__ b2, const float* __restrict__ x,
    const float* __restrict__ alpha, const int* __restrict__ list,
    const int* __restrict__ desc_e, const int* __restrict__ desc_s,
    float* __restrict__ out)
{
    const int row = blockIdx.y;
    const int e   = desc_e[row];
    if (e < 0) return;
    const int rs   = desc_s[row];
    const int wv   = threadIdx.x >> 5;
    const int lane = threadIdx.x & 31;
    const int ncol0 = (blockIdx.x * 8 + wv) * (16 * NBLK);
    const int m    = lane & 15;
    const int half = lane >> 4;
    const int kbA  = half * 8;
    const int kbB  = half * 16;

    const __bf16* aptr = Hbuf + (long)(rs + m) * FF;  // padded rows hold junk; outputs skipped
    const __bf16* wbase = w2t + (long)e * DD * FF;
    const __bf16* bcol[NBLK];
    #pragma unroll
    for (int j = 0; j < NBLK; ++j)
        bcol[j] = wbase + (long)(ncol0 + j * 16 + m) * FF;

    v8f cacc[NBLK];
    #pragma unroll
    for (int j = 0; j < NBLK; ++j) cacc[j] = {};

    #pragma unroll 2
    for (int k0 = 0; k0 < FF; k0 += 32) {
        const v16bf a = load_frag_A(aptr, k0, kbA);
        #pragma unroll
        for (int j = 0; j < NBLK; ++j) {
            const v16bf b = load_frag_B(bcol[j], k0, kbB);
            cacc[j] = __builtin_amdgcn_wmma_f32_16x16x32_bf16(
                false, a, false, b, (short)0, cacc[j], false, false);
        }
    }

    #pragma unroll
    for (int r = 0; r < 8; ++r) {
        const int M = r + half * 8;
        const int t = list[rs + M];
        if (t >= 0) {
            const float av = alpha[t];
            #pragma unroll
            for (int j = 0; j < NBLK; ++j) {
                const int  col = ncol0 + j * 16 + m;
                const long idx = (long)t * DD + col;
                out[idx] = x[idx] + av * (cacc[j][r] + b2[(long)e * DD + col]);
            }
        }
    }
}

// ---------------------------------------------------------------------------
// Host launcher
// ---------------------------------------------------------------------------
extern "C" void kernel_launch(void* const* d_in, const int* in_sizes, int n_in,
                              void* d_out, int out_size, void* d_ws, size_t ws_size,
                              hipStream_t stream)
{
    const float* x     = (const float*)d_in[0];   // [2,1024,1024]
    const float* cent  = (const float*)d_in[1];   // [8,1024]
    const float* w1    = (const float*)d_in[2];   // [8,1024,4096]
    const float* b1    = (const float*)d_in[3];   // [8,4096]
    const float* w2    = (const float*)d_in[4];   // [8,4096,1024]
    const float* b2    = (const float*)d_in[5];   // [8,1024]
    const float* gamma = (const float*)d_in[6];   // [8,1024]
    const float* beta  = (const float*)d_in[7];   // [8,1024]
    float* out = (float*)d_out;

    // workspace carving (256B aligned)
    char* ws = (char*)d_ws;
    size_t off = 0;
    auto alloc = [&](size_t bytes) -> void* {
        void* p = ws + off;
        off = (off + bytes + 255) & ~(size_t)255;
        return p;
    };
    __bf16* hbuf    = (__bf16*)alloc((size_t)TT * DD * 2);           //  4 MB
    __bf16* Hbuf    = (__bf16*)alloc((size_t)LISTPAD * FF * 2);      // 17 MB
    __bf16* w1t     = (__bf16*)alloc((size_t)NE * DD * FF * 2);      // 64 MB
    __bf16* w2t     = (__bf16*)alloc((size_t)NE * DD * FF * 2);      // 64 MB
    int*    eid     = (int*)   alloc((size_t)TT * 4);
    float*  alpha   = (float*) alloc((size_t)TT * 4);
    int*    list    = (int*)   alloc((size_t)LISTPAD * 4);
    int*    desc_e  = (int*)   alloc((size_t)MAXROWS * 4);
    int*    desc_s  = (int*)   alloc((size_t)MAXROWS * 4);
    __bf16* zerobuf = (__bf16*)alloc((size_t)FF * 2);
    (void)ws_size; (void)in_sizes; (void)n_in; (void)out_size;

    // 1) routing + LN + expert-affine (bf16 h)
    prep_kernel<<<TT, 256, 0, stream>>>(x, cent, gamma, beta, hbuf, eid, alpha);
    // 2) deterministic token lists + tile descriptors
    scan_kernel<<<1, 256, 0, stream>>>(eid, list, desc_e, desc_s, zerobuf);
    // 3) weights -> bf16 [N][K] (transposed) for contiguous WMMA B fragments
    tconv_kernel<<<dim3(FF / 32, DD / 32, NE), 256, 0, stream>>>(w1, w1t, DD, FF);
    tconv_kernel<<<dim3(DD / 32, FF / 32, NE), 256, 0, stream>>>(w2, w2t, FF, DD);
    // 4) H = relu(h @ w1 + b1)
    gemm1_kernel<<<dim3(FF / (16 * NBLK * 8), MAXROWS), 256, 0, stream>>>(
        hbuf, w1t, b1, list, desc_e, desc_s, zerobuf, Hbuf);
    // 5) out = x + alpha * (H @ w2 + b2)
    gemm2_kernel<<<dim3(DD / (16 * NBLK * 8), MAXROWS), 256, 0, stream>>>(
        Hbuf, w2t, b2, x, alpha, list, desc_e, desc_s, out);
}
